// Expert_13305808683473
// MI455X (gfx1250) — matrix operand order repaired and checked by reading, not compiled, not run on hardware
//
#include <hip/hip_runtime.h>
#include <math.h>

// ---------------------------------------------------------------------------
// Types for CDNA5 WMMA (wave32): v_wmma_f32_16x16x32_bf16
// ---------------------------------------------------------------------------
typedef __attribute__((ext_vector_type(16))) __bf16 v16bf;
typedef __attribute__((ext_vector_type(8)))  float  v8f;

union FragU {
    v16bf v;
    unsigned int u[8];
};

// Native converts (clang lowers to v_cvt_* on gfx1250, RNE). Storage stays
// unsigned short; no __bf16 arithmetic anywhere.
__device__ __forceinline__ unsigned short f2bf(float x) {
    union { __bf16 h; unsigned short u; } v;
    v.h = (__bf16)x;
    return v.u;
}
__device__ __forceinline__ float bf2f(unsigned short h) {
    union { __bf16 h; unsigned short u; } v;
    v.u = h;
    return (float)v.h;
}

// Async global->LDS copy of 16 bytes per lane (ASYNCcnt-tracked, ISA ch.8/10).
__device__ __forceinline__ void async_copy16(unsigned int lds_byte_addr,
                                             unsigned long long gaddr) {
    asm volatile("global_load_async_to_lds_b128 %0, %1, off"
                 :: "v"(lds_byte_addr), "v"(gaddr)
                 : "memory");
}
__device__ __forceinline__ void wait_async0() {
    asm volatile("s_wait_asynccnt 0" ::: "memory");
}

// B-matrix fragment (32x16 bf16) from transposed LDS tile sX[n][k] (padded):
// lane half h holds K=8h..8h+7 then K=16+8h..16+8h+7, n = lane&15.
__device__ __forceinline__ v16bf load_b_frag(const unsigned short* rowp, int half) {
    FragU f;
#pragma unroll
    for (int p = 0; p < 8; ++p) {
        int kb = (half << 3) + ((p >> 2) << 4) + ((p & 3) << 1);
        f.u[p] = *(const unsigned int*)(rowp + kb);
    }
    return f.v;
}

// ---------------------------------------------------------------------------
// One-shot weight conversion f32 -> bf16 bits.
// ---------------------------------------------------------------------------
__global__ void f32_to_bf16_kernel(const float* __restrict__ src,
                                   unsigned short* __restrict__ dst, int n) {
    int i = blockIdx.x * blockDim.x + threadIdx.x;
    if (i < n) dst[i] = f2bf(src[i]);
}

// ---------------------------------------------------------------------------
// Fused GEMM  Y[b,o,n] = sum_c W[o,c] * act(X[b,c,n]) (+bias), where act is
// identity (layer 1) or relu(scale[c]*x + shift[c]) (previous BN folded in).
// Accumulates per-channel sum/sumsq of the f32 outputs for this layer's BN.
//
// Block: 256 threads = 8 waves; block tile 128(o) x 64(n); K-step 32.
// Wave tile 32x32 -> 4 x v_wmma_f32_16x16x32_bf16.
// W tiles: fragment-major LDS layout, filled by global_load_async_to_lds_b128
//          (weights pre-converted to bf16). Lane stride 48B => conflict-free
//          v16bf reads, 16B-aligned async B128 writes.
// X tiles: global loads to regs -> BN+ReLU+cvt -> transposed padded LDS.
// Double-buffered LDS, software-pipelined against the WMMAs.
// ---------------------------------------------------------------------------
template<int IC, int OC, bool IN_F32, bool APPLY_BN>
__global__ __launch_bounds__(256)
void gemm_bn_stats(const void* __restrict__ Xin,
                   const unsigned short* __restrict__ Wbf,   // (OC,IC) bf16
                   const float* __restrict__ bias,
                   const float* __restrict__ scale_in,
                   const float* __restrict__ shift_in,
                   unsigned short* __restrict__ Yout,        // bf16 bits
                   float* __restrict__ gsum,
                   float* __restrict__ gsq)
{
    constexpr int N  = 256;
    constexpr int BM = 128, BN = 64, BK = 32;
    constexpr int LDT = BK + 2;          // X tile: 34 -> 17-DWORD rows
    constexpr int WSLOT = 24;            // 48 bytes per lane slot (u16 units)
    constexpr int WSUB  = 32 * WSLOT;    // one 16x32 sub-tile = 768 u16
    constexpr int KT = IC / BK;

    __shared__ __align__(64) unsigned short sWf[2][8 * WSUB]; // 2 x 12 KB
    __shared__ unsigned short sX[2][BN * LDT];                // 2 x 4.25 KB
    __shared__ float sSum[BM];
    __shared__ float sSq[BM];

    const int b    = blockIdx.z;
    const int o0   = blockIdx.y * BM;
    const int n0   = blockIdx.x * BN;
    const int t    = threadIdx.x;
    const int lane = t & 31;
    const int wv   = t >> 5;
    const int half = lane >> 4;
    const int ln   = lane & 15;
    const int wo   = (wv >> 1) * 32;     // wave o-offset in block tile
    const int wn   = (wv & 1) * 32;      // wave n-offset in block tile

    if (t < BM) { sSum[t] = 0.f; sSq[t] = 0.f; }

    const float*          Xf = (const float*)Xin;
    const unsigned short* Xh = (const unsigned short*)Xin;

    struct XStage {
        union { float f[8]; uint4 raw; float4 f4[2]; };
        float sc, sh;
    };

    // ---- pipeline helper lambdas -----------------------------------------
    auto stageW = [&](int bi, int k0) {
        // thread t fills lane slot l of sub-tile s with its two 16B chunks
        int s = t >> 5, l = t & 31;
        int o = o0 + s * 16 + (l & 15);
        int h = l >> 4;
        const unsigned short* g0 = Wbf + (size_t)o * IC + k0 + h * 8;
        unsigned int lds0 = (unsigned int)(size_t)&sWf[bi][0]
                          + (unsigned int)(s * (WSUB * 2) + l * 48);
        async_copy16(lds0,      (unsigned long long)(size_t)g0);
        async_copy16(lds0 + 16, (unsigned long long)(size_t)(g0 + 16));
    };
    auto loadX = [&](XStage& xs, int k0) {
        int kk = t >> 3;                  // k within tile (0..31)
        int ng = (t & 7) * 8;             // n group of 8
        int c  = k0 + kk;
        if (APPLY_BN) { xs.sc = scale_in[c]; xs.sh = shift_in[c]; }
        size_t base = ((size_t)b * IC + c) * N + n0 + ng;
        if (IN_F32) {
            xs.f4[0] = *(const float4*)(Xf + base);
            xs.f4[1] = *(const float4*)(Xf + base + 4);
        } else {
            xs.raw = *(const uint4*)(Xh + base);
        }
    };
    auto storeX = [&](int bi, const XStage& xs) {
        int kk = t >> 3;
        int ng = (t & 7) * 8;
        unsigned short* dst = &sX[bi][kk];
#pragma unroll
        for (int j = 0; j < 8; ++j) {
            float v = IN_F32 ? xs.f[j]
                             : bf2f(((const unsigned short*)&xs.raw)[j]);
            if (APPLY_BN) {
                v = v * xs.sc + xs.sh;
                v = v > 0.f ? v : 0.f;
            }
            dst[(ng + j) * LDT] = f2bf(v);
        }
    };

    // ---- software pipeline ------------------------------------------------
    v8f acc[2][2] = {};
    XStage xs;
    int buf = 0;

    stageW(0, 0);
    loadX(xs, 0);
    storeX(0, xs);
    wait_async0();
    __syncthreads();

    const int s0 = (wv >> 1) * 2;        // wave's first W sub-tile index
    for (int kt = 0; kt < KT; ++kt) {
        const int nb = buf ^ 1;
        const bool more = (kt + 1 < KT);
        if (more) {
            stageW(nb, (kt + 1) * BK);   // async engine fills next W tile
            loadX(xs, (kt + 1) * BK);    // next X tile into registers
        }
        // compute current tile
        v16bf afrag[2], bfrag[2];
#pragma unroll
        for (int oi = 0; oi < 2; ++oi)
            afrag[oi] = *(const v16bf*)&sWf[buf][(s0 + oi) * WSUB + lane * WSLOT];
#pragma unroll
        for (int ni = 0; ni < 2; ++ni)
            bfrag[ni] = load_b_frag(&sX[buf][(wn + ni * 16 + ln) * LDT], half);
#pragma unroll
        for (int oi = 0; oi < 2; ++oi)
#pragma unroll
            for (int ni = 0; ni < 2; ++ni)
                acc[oi][ni] = __builtin_amdgcn_wmma_f32_16x16x32_bf16(
                    false, afrag[oi], false, bfrag[ni],
                    (short)0, acc[oi][ni], false, false);
        if (more) {
            storeX(nb, xs);              // transform + store next X tile
            wait_async0();               // next W tile resident
        }
        __syncthreads();
        buf = nb;
    }

    // ---- epilogue: +bias, bf16 store, per-channel stats -------------------
#pragma unroll
    for (int oi = 0; oi < 2; ++oi) {
#pragma unroll
        for (int r = 0; r < 8; ++r) {
            int ml = wo + oi * 16 + (half << 3) + r;   // local o (0..127)
            int o  = o0 + ml;
            float bi = bias[o];
            float s = 0.f, s2 = 0.f;
#pragma unroll
            for (int ni = 0; ni < 2; ++ni) {
                float y = acc[oi][ni][r] + bi;
                int n = n0 + wn + ni * 16 + ln;
                Yout[((size_t)b * OC + o) * N + n] = f2bf(y);
                s  += y;
                s2 += y * y;
            }
            atomicAdd(&sSum[ml], s);
            atomicAdd(&sSq[ml], s2);
        }
    }
    __syncthreads();
    if (t < BM) {
        atomicAdd(&gsum[o0 + t], sSum[t]);
        atomicAdd(&gsq[o0 + t],  sSq[t]);
    }
}

// ---------------------------------------------------------------------------
// Per-channel BN finalize: (sum,sumsq) -> (scale, shift)
// ---------------------------------------------------------------------------
__global__ void bn_finalize(const float* __restrict__ gsum,
                            const float* __restrict__ gsq,
                            const float* __restrict__ gamma,
                            const float* __restrict__ beta,
                            float* __restrict__ scale,
                            float* __restrict__ shift,
                            int C, float invCount)
{
    int c = blockIdx.x * blockDim.x + threadIdx.x;
    if (c >= C) return;
    float mean = gsum[c] * invCount;
    float var  = gsq[c] * invCount - mean * mean;
    float rs   = rsqrtf(var + 1e-5f);
    float sc   = gamma[c] * rs;
    scale[c] = sc;
    shift[c] = beta[c] - mean * sc;
}

// ---------------------------------------------------------------------------
// Final: logits (128-dot) -> sigmoid weights -> per-batch 3x3 weighted LS.
// weights = 0.01 + sigmoid(.) > 0.001 always, so the reference's "valid"
// branch always selects the computed weights. One block per batch.
// ---------------------------------------------------------------------------
__global__ __launch_bounds__(256)
void final_fit(const unsigned short* __restrict__ Y3,   // (B,128,256) bf16
               const float* __restrict__ scale3,
               const float* __restrict__ shift3,
               const float* __restrict__ w4,
               const float* __restrict__ b4,
               const float* __restrict__ points,        // (B,3,256)
               float* __restrict__ out)
{
    constexpr int N = 256, C = 128;
    __shared__ float sw4[C], ssc[C], ssh[C];
    __shared__ float red[9];

    int b = blockIdx.x;
    int n = threadIdx.x;
    if (n < C) { sw4[n] = w4[n]; ssc[n] = scale3[n]; ssh[n] = shift3[n]; }
    if (n < 9) red[n] = 0.f;
    __syncthreads();

    float logit = b4[0];
    size_t base = (size_t)b * C * N + n;
#pragma unroll 4
    for (int c = 0; c < C; ++c) {
        float v = bf2f(Y3[base + (size_t)c * N]) * ssc[c] + ssh[c];
        v = v > 0.f ? v : 0.f;
        logit += sw4[c] * v;
    }
    float wgt = 0.01f + 1.f / (1.f + __expf(-logit));
    out[3072 + b * N + n] = wgt;                        // weights

    const float* pb = points + (size_t)b * 3 * N;
    float px = pb[n], py = pb[N + n], pz = pb[2 * N + n];
    atomicAdd(&red[0], wgt * px * px);
    atomicAdd(&red[1], wgt * px * py);
    atomicAdd(&red[2], wgt * px);
    atomicAdd(&red[3], wgt * py * py);
    atomicAdd(&red[4], wgt * py);
    atomicAdd(&red[5], wgt);
    atomicAdd(&red[6], wgt * px * pz);
    atomicAdd(&red[7], wgt * py * pz);
    atomicAdd(&red[8], wgt * pz);
    __syncthreads();

    if (n == 0) {
        float M00 = red[0], M01 = red[1], M02 = red[2];
        float M11 = red[3], M12 = red[4], M22 = red[5];
        float Y0 = red[6], Y1v = red[7], Y2v = red[8];
        float c00 = M11 * M22 - M12 * M12;
        float c01 = M12 * M02 - M01 * M22;
        float c02 = M01 * M12 - M11 * M02;
        float c11 = M00 * M22 - M02 * M02;
        float c12 = M01 * M02 - M00 * M12;
        float c22 = M00 * M11 - M01 * M01;
        float det = M00 * c00 + M01 * c01 + M02 * c02;
        float inv = 1.f / det;
        float b0 = (c00 * Y0 + c01 * Y1v + c02 * Y2v) * inv;
        float b1 = (c01 * Y0 + c11 * Y1v + c12 * Y2v) * inv;
        float b2 = (c02 * Y0 + c12 * Y1v + c22 * Y2v) * inv;
        float nx = -b0, ny = -b1, nz = 1.f;
        float nrm = rsqrtf(nx * nx + ny * ny + nz * nz);
        out[b * 3 + 0] = nx * nrm;                      // normal
        out[b * 3 + 1] = ny * nrm;
        out[b * 3 + 2] = nz * nrm;
        out[1536 + b * 3 + 0] = b0;                     // beta
        out[1536 + b * 3 + 1] = b1;
        out[1536 + b * 3 + 2] = b2;
    }
}

// ---------------------------------------------------------------------------
// Launch
// ---------------------------------------------------------------------------
extern "C" void kernel_launch(void* const* d_in, const int* in_sizes, int n_in,
                              void* d_out, int out_size, void* d_ws, size_t ws_size,
                              hipStream_t stream)
{
    (void)in_sizes; (void)n_in; (void)out_size; (void)ws_size;

    const float* points = (const float*)d_in[0];
    const float* x      = (const float*)d_in[1];
    // d_in[2] n_effective_points: unused by the reference math
    const float* w1  = (const float*)d_in[3];
    const float* b1  = (const float*)d_in[4];
    const float* g1  = (const float*)d_in[5];
    const float* be1 = (const float*)d_in[6];
    const float* w2  = (const float*)d_in[7];
    const float* b2  = (const float*)d_in[8];
    const float* g2  = (const float*)d_in[9];
    const float* be2 = (const float*)d_in[10];
    const float* w3  = (const float*)d_in[11];
    const float* b3  = (const float*)d_in[12];
    const float* g3  = (const float*)d_in[13];
    const float* be3 = (const float*)d_in[14];
    const float* w4  = (const float*)d_in[15];
    const float* b4  = (const float*)d_in[16];
    float* out = (float*)d_out;

    char* ws = (char*)d_ws;
    unsigned short* Y1 = (unsigned short*)(ws);                       // 128 MB
    unsigned short* Y2 = (unsigned short*)(ws + ((size_t)128 << 20)); // 64 MB
    unsigned short* Y3 = (unsigned short*)(ws);                       // reuse Y1
    float* stats = (float*)(ws + ((size_t)192 << 20));
    float* sum1 = stats;        float* sq1 = sum1 + 512;              // zeroed
    float* sum2 = sq1 + 512;    float* sq2 = sum2 + 256;
    float* sum3 = sq2 + 256;    float* sq3 = sum3 + 128;
    float* sc1  = sq3 + 128;    float* sh1 = sc1 + 512;               // computed
    float* sc2  = sh1 + 512;    float* sh2 = sc2 + 256;
    float* sc3  = sh2 + 256;    float* sh3 = sc3 + 128;
    unsigned short* wbf1 = (unsigned short*)(ws + ((size_t)192 << 20) + (16 << 10));
    unsigned short* wbf2 = wbf1 + 512 * 1024;
    unsigned short* wbf3 = wbf2 + 256 * 512;

    hipMemsetAsync(stats, 0, (size_t)(2 * (512 + 256 + 128)) * sizeof(float), stream);

    // Pre-convert weights to bf16 so W tiles can be DMA'd (async -> LDS).
    f32_to_bf16_kernel<<<(512 * 1024) / 256, 256, 0, stream>>>(w1, wbf1, 512 * 1024);
    f32_to_bf16_kernel<<<(256 * 512)  / 256, 256, 0, stream>>>(w2, wbf2, 256 * 512);
    f32_to_bf16_kernel<<<(128 * 256)  / 256, 256, 0, stream>>>(w3, wbf3, 128 * 256);

    const float invCnt = 1.f / (512.f * 256.f);

    // Layer 1: 1024 -> 512 (input f32, no BN on input)
    gemm_bn_stats<1024, 512, true, false>
        <<<dim3(4, 4, 512), 256, 0, stream>>>(x, wbf1, b1, nullptr, nullptr, Y1, sum1, sq1);
    bn_finalize<<<2, 256, 0, stream>>>(sum1, sq1, g1, be1, sc1, sh1, 512, invCnt);

    // Layer 2: 512 -> 256 (input bf16, BN1+ReLU folded into load)
    gemm_bn_stats<512, 256, false, true>
        <<<dim3(4, 2, 512), 256, 0, stream>>>(Y1, wbf2, b2, sc1, sh1, Y2, sum2, sq2);
    bn_finalize<<<1, 256, 0, stream>>>(sum2, sq2, g2, be2, sc2, sh2, 256, invCnt);

    // Layer 3: 256 -> 128 (input bf16, BN2+ReLU folded into load)
    gemm_bn_stats<256, 128, false, true>
        <<<dim3(4, 1, 512), 256, 0, stream>>>(Y2, wbf3, b3, sc2, sh2, Y3, sum3, sq3);
    bn_finalize<<<1, 128, 0, stream>>>(sum3, sq3, g3, be3, sc3, sh3, 128, invCnt);

    // Final: logits + sigmoid weights + per-batch 3x3 WLS solve
    final_fit<<<512, 256, 0, stream>>>(Y3, sc3, sh3, w4, b4, points, out);
}
